// PGWCrossAttention_53944789237785
// MI455X (gfx1250) — matrix operand order repaired and checked by reading, not compiled, not run on hardware
//
#include <hip/hip_runtime.h>
#include <hip/hip_bf16.h>
#include <math.h>

#define BB  8
#define NN  2048
#define MM  2048
#define DIN 512
#define HH  256
#define AW  4   // waves per block, attention kernels
#define MW  4   // waves per block, mlp

typedef __bf16 bf16;
typedef bf16  v16bf __attribute__((ext_vector_type(16)));
typedef float v8f   __attribute__((ext_vector_type(8)));

__device__ __forceinline__ bf16 f2bf(float f) {
  union { float f; unsigned int u; } c; c.f = f;
  unsigned int r = c.u + 0x7FFFu + ((c.u >> 16) & 1u);
  unsigned short h = (unsigned short)(r >> 16);
  return __builtin_bit_cast(bf16, h);
}

union V16U { v16bf v; uint4 q[2]; };
union P8U  { bf16 h[8]; uint4 q; };

// A fragment (16x32, M x K) from row-major memory, ld in elements.
// Lanes 0-15: row M=lane, K {0..7,16..23}; lanes 16-31: K {8..15,24..31}.
__device__ __forceinline__ v16bf load_A_bf(const bf16* base, int ld) {
  int lane = threadIdx.x & 31;
  int rr = lane & 15, hh = lane >> 4;
  const bf16* p = base + (size_t)rr * ld + 8 * hh;
  V16U u;
  u.q[0] = *(const uint4*)p;          // K = 8h .. 8h+7
  u.q[1] = *(const uint4*)(p + 16);   // K = 16+8h .. 16+8h+7
  return u.v;
}

// B fragment (32x16, K x N) where memory is N x K row-major (per-lane contiguous K).
__device__ __forceinline__ v16bf load_B_rows(const bf16* base, int ld) {
  int lane = threadIdx.x & 31;
  int nn = lane & 15, hh = lane >> 4;
  const bf16* p = base + (size_t)nn * ld + 8 * hh;  // K {8h..8h+7, 16+8h..16+8h+7}
  V16U u;
  u.q[0] = *(const uint4*)p;
  u.q[1] = *(const uint4*)(p + 16);
  return u.v;
}

__device__ __forceinline__ v8f wmma_bf16(v16bf a, v16bf b, v8f c) {
  return __builtin_amdgcn_wmma_f32_16x16x32_bf16(false, a, false, b, (short)0, c, false, false);
}

__device__ __forceinline__ float redmax16(float v) {
  v = fmaxf(v, __shfl_xor(v, 1, 32));
  v = fmaxf(v, __shfl_xor(v, 2, 32));
  v = fmaxf(v, __shfl_xor(v, 4, 32));
  v = fmaxf(v, __shfl_xor(v, 8, 32));
  return v;
}
__device__ __forceinline__ float redsum16(float v) {
  v += __shfl_xor(v, 1, 32);
  v += __shfl_xor(v, 2, 32);
  v += __shfl_xor(v, 4, 32);
  v += __shfl_xor(v, 8, 32);
  return v;
}
__device__ __forceinline__ float redsum32(float v) {
  v += __shfl_xor(v, 1, 32);
  v += __shfl_xor(v, 2, 32);
  v += __shfl_xor(v, 4, 32);
  v += __shfl_xor(v, 8, 32);
  v += __shfl_xor(v, 16, 32);
  return v;
}

// ---------------- Prep: fp32 -> bf16 elementwise --------------------------------------
__global__ __launch_bounds__(256) void cvt_bf16_kernel(
    const float* __restrict__ x, bf16* __restrict__ y, size_t n) {
  size_t i = ((size_t)blockIdx.x * 256 + threadIdx.x) * 4;
  if (i + 3 < n) {
    float4 v = *(const float4*)(x + i);
    y[i + 0] = f2bf(v.x); y[i + 1] = f2bf(v.y);
    y[i + 2] = f2bf(v.z); y[i + 3] = f2bf(v.w);
  }
}

// ---------------- Prep: W [K,N] fp32 -> WT [N,K] bf16 ----------------------------------
__global__ __launch_bounds__(256) void transpose_w_kernel(
    const float* __restrict__ W, bf16* __restrict__ WT, int K, int N) {
  int idx = blockIdx.x * 256 + threadIdx.x;
  if (idx < K * N) {
    int k = idx / N, n = idx - k * N;
    WT[(size_t)n * K + k] = f2bf(W[idx]);
  }
}

// ---------------- Projections, q side: qs, qe (+ row |q_elec|^2) ------------------------
__global__ __launch_bounds__(256) void proj2_kernel(
    const bf16* __restrict__ X,
    const bf16* __restrict__ WTa, const float* __restrict__ ba,
    const bf16* __restrict__ WTb, const float* __restrict__ bb,
    bf16* __restrict__ Ya, bf16* __restrict__ Yb, float* __restrict__ sqb) {
  __shared__ float sq_sh[16];
  int wave = threadIdx.x >> 5, lane = threadIdx.x & 31;
  int nn = lane & 15, hh = lane >> 4;
  int r0 = blockIdx.x * 16, c0 = wave * 32;
  if (threadIdx.x < 16) sq_sh[threadIdx.x] = 0.f;
  __syncthreads();
  v8f a0 = {}, a1 = {}, b0 = {}, b1 = {};
  for (int k0 = 0; k0 < DIN; k0 += 32) {
    v16bf af = load_A_bf(X + (size_t)r0 * DIN + k0, DIN);
    a0 = wmma_bf16(af, load_B_rows(WTa + (size_t)c0 * DIN + k0, DIN), a0);
    a1 = wmma_bf16(af, load_B_rows(WTa + (size_t)(c0 + 16) * DIN + k0, DIN), a1);
    b0 = wmma_bf16(af, load_B_rows(WTb + (size_t)c0 * DIN + k0, DIN), b0);
    b1 = wmma_bf16(af, load_B_rows(WTb + (size_t)(c0 + 16) * DIN + k0, DIN), b1);
  }
  float ba0 = ba[c0 + nn], ba1 = ba[c0 + 16 + nn];
  float bb0 = bb[c0 + nn], bb1 = bb[c0 + 16 + nn];
#pragma unroll
  for (int j = 0; j < 8; ++j) {
    a0[j] += ba0; a1[j] += ba1; b0[j] += bb0; b1[j] += bb1;
  }
#pragma unroll
  for (int j = 0; j < 8; ++j) {
    size_t row = (size_t)(r0 + 8 * hh + j);
    Ya[row * HH + c0 + nn]      = f2bf(a0[j]);
    Ya[row * HH + c0 + 16 + nn] = f2bf(a1[j]);
    Yb[row * HH + c0 + nn]      = f2bf(b0[j]);
    Yb[row * HH + c0 + 16 + nn] = f2bf(b1[j]);
  }
#pragma unroll
  for (int j = 0; j < 8; ++j) {
    float v = b0[j] * b0[j] + b1[j] * b1[j];
    v = redsum16(v);
    if (nn == 0) atomicAdd(&sq_sh[8 * hh + j], v);
  }
  __syncthreads();
  if (threadIdx.x < 16) sqb[r0 + threadIdx.x] = sq_sh[threadIdx.x];
}

// ---------------- Projections, kv side: ks, ke (+|k_elec|^2), vs^T, ve^T ---------------
__global__ __launch_bounds__(256) void proj4_kernel(
    const bf16* __restrict__ X,
    const bf16* __restrict__ WTks, const float* __restrict__ bks,
    const bf16* __restrict__ WTke, const float* __restrict__ bke,
    const bf16* __restrict__ WTvs, const float* __restrict__ bvs,
    const bf16* __restrict__ WTve, const float* __restrict__ bve,
    bf16* __restrict__ Yks, bf16* __restrict__ Yke,
    bf16* __restrict__ vTs, bf16* __restrict__ vTe, float* __restrict__ k2) {
  __shared__ float sq_sh[16];
  int wave = threadIdx.x >> 5, lane = threadIdx.x & 31;
  int nn = lane & 15, hh = lane >> 4;
  int r0 = blockIdx.x * 16, c0 = wave * 32;
  int b = r0 / MM, m0 = r0 - b * MM;
  if (threadIdx.x < 16) sq_sh[threadIdx.x] = 0.f;
  __syncthreads();
  v8f ks0 = {}, ks1 = {}, ke0 = {}, ke1 = {};
  v8f vs0 = {}, vs1 = {}, ve0 = {}, ve1 = {};
  for (int k0 = 0; k0 < DIN; k0 += 32) {
    v16bf af = load_A_bf(X + (size_t)r0 * DIN + k0, DIN);
    ks0 = wmma_bf16(af, load_B_rows(WTks + (size_t)c0 * DIN + k0, DIN), ks0);
    ks1 = wmma_bf16(af, load_B_rows(WTks + (size_t)(c0 + 16) * DIN + k0, DIN), ks1);
    ke0 = wmma_bf16(af, load_B_rows(WTke + (size_t)c0 * DIN + k0, DIN), ke0);
    ke1 = wmma_bf16(af, load_B_rows(WTke + (size_t)(c0 + 16) * DIN + k0, DIN), ke1);
    vs0 = wmma_bf16(af, load_B_rows(WTvs + (size_t)c0 * DIN + k0, DIN), vs0);
    vs1 = wmma_bf16(af, load_B_rows(WTvs + (size_t)(c0 + 16) * DIN + k0, DIN), vs1);
    ve0 = wmma_bf16(af, load_B_rows(WTve + (size_t)c0 * DIN + k0, DIN), ve0);
    ve1 = wmma_bf16(af, load_B_rows(WTve + (size_t)(c0 + 16) * DIN + k0, DIN), ve1);
  }
  float bk0 = bks[c0 + nn], bk1 = bks[c0 + 16 + nn];
  float be0 = bke[c0 + nn], be1 = bke[c0 + 16 + nn];
  float bv0 = bvs[c0 + nn], bv1 = bvs[c0 + 16 + nn];
  float bw0 = bve[c0 + nn], bw1 = bve[c0 + 16 + nn];
#pragma unroll
  for (int j = 0; j < 8; ++j) {
    ks0[j] += bk0; ks1[j] += bk1; ke0[j] += be0; ke1[j] += be1;
    vs0[j] += bv0; vs1[j] += bv1; ve0[j] += bw0; ve1[j] += bw1;
  }
  // row-major stores for ks / ke
#pragma unroll
  for (int j = 0; j < 8; ++j) {
    size_t row = (size_t)(r0 + 8 * hh + j);
    Yks[row * HH + c0 + nn]      = f2bf(ks0[j]);
    Yks[row * HH + c0 + 16 + nn] = f2bf(ks1[j]);
    Yke[row * HH + c0 + nn]      = f2bf(ke0[j]);
    Yke[row * HH + c0 + 16 + nn] = f2bf(ke1[j]);
  }
  // transposed [B,H,M] stores for vs / ve: lane holds 8 consecutive m for one column
  {
    P8U u;
    size_t mbase = (size_t)m0 + 8 * hh;
    size_t base = ((size_t)b * HH) * MM;
#pragma unroll
    for (int j = 0; j < 8; ++j) u.h[j] = f2bf(vs0[j]);
    *(uint4*)(vTs + base + (size_t)(c0 + nn) * MM + mbase) = u.q;
#pragma unroll
    for (int j = 0; j < 8; ++j) u.h[j] = f2bf(vs1[j]);
    *(uint4*)(vTs + base + (size_t)(c0 + 16 + nn) * MM + mbase) = u.q;
#pragma unroll
    for (int j = 0; j < 8; ++j) u.h[j] = f2bf(ve0[j]);
    *(uint4*)(vTe + base + (size_t)(c0 + nn) * MM + mbase) = u.q;
#pragma unroll
    for (int j = 0; j < 8; ++j) u.h[j] = f2bf(ve1[j]);
    *(uint4*)(vTe + base + (size_t)(c0 + 16 + nn) * MM + mbase) = u.q;
  }
#pragma unroll
  for (int j = 0; j < 8; ++j) {
    float v = ke0[j] * ke0[j] + ke1[j] * ke1[j];
    v = redsum16(v);
    if (nn == 0) atomicAdd(&sq_sh[8 * hh + j], v);
  }
  __syncthreads();
  if (threadIdx.x < 16) k2[r0 + threadIdx.x] = sq_sh[threadIdx.x];
}

// ---------------- Pass 1: per-batch sum of electronic distances -------------------------
__global__ __launch_bounds__(AW * 32) void dist_sum_kernel(
    const bf16* __restrict__ qe, const bf16* __restrict__ ke,
    const float* __restrict__ q2, const float* __restrict__ k2,
    float* __restrict__ dist_sum) {
  int wave = threadIdx.x >> 5, lane = threadIdx.x & 31;
  int nn = lane & 15, hh = lane >> 4;
  int b = blockIdx.y;
  int n0 = (blockIdx.x * AW + wave) * 16;
  const bf16* qe_b = qe + ((size_t)b * NN + n0) * HH;
  const bf16* ke_b = ke + (size_t)b * MM * HH;
  const float* q2_b = q2 + (size_t)b * NN + n0;
  const float* k2_b = k2 + (size_t)b * MM;
  v16bf aq[8];
#pragma unroll
  for (int t = 0; t < 8; ++t) aq[t] = load_A_bf(qe_b + 32 * t, HH);
  float q2r[8];
#pragma unroll
  for (int j = 0; j < 8; ++j) q2r[j] = q2_b[8 * hh + j];
  float acc = 0.f;
  for (int m0 = 0; m0 < MM; m0 += 16) {
    if (m0 + 16 < MM)
      __builtin_prefetch((const void*)(ke_b + (size_t)(m0 + 16) * HH), 0, 1);
    v8f e = {};
#pragma unroll
    for (int t = 0; t < 8; ++t)
      e = wmma_bf16(aq[t], load_B_rows(ke_b + (size_t)m0 * HH + 32 * t, HH), e);
    float k2v = k2_b[m0 + nn];
#pragma unroll
    for (int j = 0; j < 8; ++j)
      acc += sqrtf(fmaxf(q2r[j] + k2v - 2.f * e[j], 1e-12f));
  }
  acc = redsum32(acc);
  if (lane == 0) atomicAdd(&dist_sum[b], acc);
}

// ---------------- Pass 2: fused flash attention -> ctx_s, ctx_e -------------------------
__global__ __launch_bounds__(AW * 32) void attn_kernel(
    const bf16* __restrict__ qs, const bf16* __restrict__ qe,
    const bf16* __restrict__ ks, const bf16* __restrict__ ke,
    const bf16* __restrict__ vTs, const bf16* __restrict__ vTe,
    const float* __restrict__ q2, const float* __restrict__ k2,
    const float* __restrict__ pi, const float* __restrict__ dist_sum,
    const float* __restrict__ gamma_p, const float* __restrict__ ew_p,
    float* __restrict__ ctx_s, float* __restrict__ ctx_e) {
  __shared__ bf16 Pbuf[AW][16 * 32];
  int wave = threadIdx.x >> 5, lane = threadIdx.x & 31;
  int nn = lane & 15, hh = lane >> 4;
  int b = blockIdx.y;
  int n0 = (blockIdx.x * AW + wave) * 16;
  float escale = fmaxf(dist_sum[b] * (1.0f / ((float)NN * (float)MM)), 1e-4f);
  float cd = -ew_p[0] / escale;       // coefficient on distance
  float gamma = gamma_p[0];
  const float inv_sq = 0.0625f;       // 1/sqrt(256)
  const bf16* qs_b = qs + ((size_t)b * NN + n0) * HH;
  const bf16* qe_b = qe + ((size_t)b * NN + n0) * HH;
  const bf16* ks_b = ks + (size_t)b * MM * HH;
  const bf16* ke_b = ke + (size_t)b * MM * HH;
  const bf16* vTs_b = vTs + (size_t)b * HH * MM;
  const bf16* vTe_b = vTe + (size_t)b * HH * MM;
  const float* k2_b = k2 + (size_t)b * MM;

  v16bf aqs[8], aqe[8];
#pragma unroll
  for (int t = 0; t < 8; ++t) {
    aqs[t] = load_A_bf(qs_b + 32 * t, HH);
    aqe[t] = load_A_bf(qe_b + 32 * t, HH);
  }
  float q2r[8];
#pragma unroll
  for (int j = 0; j < 8; ++j) q2r[j] = q2[(size_t)b * NN + n0 + 8 * hh + j];

  float mi[8], li[8];
#pragma unroll
  for (int j = 0; j < 8; ++j) { mi[j] = -1e30f; li[j] = 0.f; }
  v8f o[32];
#pragma unroll
  for (int t = 0; t < 32; ++t) o[t] = (v8f){};

  for (int m0 = 0; m0 < MM; m0 += 32) {
    if (m0 + 32 < MM) {
      __builtin_prefetch((const void*)(ks_b + (size_t)(m0 + 32) * HH), 0, 1);
      __builtin_prefetch((const void*)(ke_b + (size_t)(m0 + 32) * HH), 0, 1);
    }
    v8f s0 = {}, s1 = {}, e0 = {}, e1 = {};
#pragma unroll
    for (int t = 0; t < 8; ++t) {
      s0 = wmma_bf16(aqs[t], load_B_rows(ks_b + (size_t)m0 * HH + 32 * t, HH), s0);
      e0 = wmma_bf16(aqe[t], load_B_rows(ke_b + (size_t)m0 * HH + 32 * t, HH), e0);
      s1 = wmma_bf16(aqs[t], load_B_rows(ks_b + (size_t)(m0 + 16) * HH + 32 * t, HH), s1);
      e1 = wmma_bf16(aqe[t], load_B_rows(ke_b + (size_t)(m0 + 16) * HH + 32 * t, HH), e1);
    }
    float k2v0 = k2_b[m0 + nn], k2v1 = k2_b[m0 + 16 + nn];
    const float* pi_b = pi + ((size_t)b * NN + n0) * MM + m0;
    float lg0[8], lg1[8];
#pragma unroll
    for (int j = 0; j < 8; ++j) {
      const float* pr = pi_b + (size_t)(8 * hh + j) * MM;
      float p0 = pr[nn], p1 = pr[16 + nn];
      float d0 = sqrtf(fmaxf(q2r[j] + k2v0 - 2.f * e0[j], 1e-12f));
      float d1 = sqrtf(fmaxf(q2r[j] + k2v1 - 2.f * e1[j], 1e-12f));
      lg0[j] = s0[j] * inv_sq + cd * d0 + gamma * __logf(fmaxf(p0, 1e-9f));
      lg1[j] = s1[j] * inv_sq + cd * d1 + gamma * __logf(fmaxf(p1, 1e-9f));
    }
#pragma unroll
    for (int j = 0; j < 8; ++j) {
      float cm = redmax16(fmaxf(lg0[j], lg1[j]));
      float mn = fmaxf(mi[j], cm);
      float alpha = __expf(mi[j] - mn);
      mi[j] = mn;
      float p0 = __expf(lg0[j] - mn);
      float p1 = __expf(lg1[j] - mn);
      li[j] = li[j] * alpha + redsum16(p0 + p1);
#pragma unroll
      for (int t = 0; t < 32; ++t) o[t][j] *= alpha;
      Pbuf[wave][(8 * hh + j) * 32 + nn]      = f2bf(p0);
      Pbuf[wave][(8 * hh + j) * 32 + 16 + nn] = f2bf(p1);
    }
    v16bf ap = load_A_bf(&Pbuf[wave][0], 32);
#pragma unroll
    for (int t = 0; t < 16; ++t)
      o[t] = wmma_bf16(ap, load_B_rows(vTs_b + (size_t)(16 * t) * MM + m0, MM), o[t]);
#pragma unroll
    for (int t = 0; t < 16; ++t)
      o[16 + t] = wmma_bf16(ap, load_B_rows(vTe_b + (size_t)(16 * t) * MM + m0, MM), o[16 + t]);
  }
#pragma unroll
  for (int j = 0; j < 8; ++j) {
    float inv_l = 1.0f / li[j];
    size_t row = (size_t)b * NN + n0 + 8 * hh + j;
#pragma unroll
    for (int t = 0; t < 16; ++t) {
      ctx_s[row * HH + 16 * t + nn] = o[t][j] * inv_l;
      ctx_e[row * HH + 16 * t + nn] = o[16 + t][j] * inv_l;
    }
  }
}

// ---------------- MLP: silu([cs,ce,cs-ce,cs*ce] @ W1 + b1) @ W2 + b2 --------------------
__global__ __launch_bounds__(MW * 32) void mlp_kernel(
    const float* __restrict__ cs, const float* __restrict__ ce,
    const bf16* __restrict__ W1T, const float* __restrict__ b1,
    const bf16* __restrict__ W2T, const float* __restrict__ b2,
    float* __restrict__ out) {
  __shared__ bf16 Hbuf[MW][16 * HH];
  int wave = threadIdx.x >> 5, lane = threadIdx.x & 31;
  int nn = lane & 15, hh = lane >> 4, rr = lane & 15;
  int r0 = (blockIdx.x * MW + wave) * 16;
  const float* csr = cs + (size_t)(r0 + rr) * HH;
  const float* cer = ce + (size_t)(r0 + rr) * HH;
  v8f acc[16];
#pragma unroll
  for (int t = 0; t < 16; ++t) acc[t] = (v8f){};
  for (int k0 = 0; k0 < 4 * HH; k0 += 32) {
    int seg = k0 >> 8;
    int hk = (k0 & 255) + 8 * hh;
    v16bf af;
#pragma unroll
    for (int e = 0; e < 8; ++e) {
      float a = csr[hk + e], c = cer[hk + e];
      af[e] = f2bf(seg == 0 ? a : seg == 1 ? c : seg == 2 ? (a - c) : (a * c));
    }
#pragma unroll
    for (int e = 0; e < 8; ++e) {
      float a = csr[hk + 16 + e], c = cer[hk + 16 + e];
      af[8 + e] = f2bf(seg == 0 ? a : seg == 1 ? c : seg == 2 ? (a - c) : (a * c));
    }
#pragma unroll
    for (int t = 0; t < 16; ++t)
      acc[t] = wmma_bf16(af, load_B_rows(W1T + (size_t)(16 * t) * (4 * HH) + k0, 4 * HH),
                         acc[t]);
  }
#pragma unroll
  for (int t = 0; t < 16; ++t) {
    float bv = b1[16 * t + nn];
#pragma unroll
    for (int j = 0; j < 8; ++j) {
      float x = acc[t][j] + bv;
      float s = x / (1.f + __expf(-x));
      Hbuf[wave][(8 * hh + j) * HH + 16 * t + nn] = f2bf(s);
    }
  }
  v8f o[16];
#pragma unroll
  for (int t = 0; t < 16; ++t) o[t] = (v8f){};
  for (int k0 = 0; k0 < HH; k0 += 32) {
    v16bf af = load_A_bf(&Hbuf[wave][k0], HH);
#pragma unroll
    for (int t = 0; t < 16; ++t)
      o[t] = wmma_bf16(af, load_B_rows(W2T + (size_t)(16 * t) * HH + k0, HH), o[t]);
  }
#pragma unroll
  for (int t = 0; t < 16; ++t) {
    float bv = b2[16 * t + nn];
#pragma unroll
    for (int j = 0; j < 8; ++j)
      out[(size_t)(r0 + 8 * hh + j) * HH + 16 * t + nn] = o[t][j] + bv;
  }
}

extern "C" void kernel_launch(void* const* d_in, const int* in_sizes, int n_in,
                              void* d_out, int out_size, void* d_ws, size_t ws_size,
                              hipStream_t stream) {
  const float* q_fp  = (const float*)d_in[0];
  const float* v_ret = (const float*)d_in[1];
  const float* pi    = (const float*)d_in[2];
  const float* Wqs = (const float*)d_in[3];  const float* bqs = (const float*)d_in[4];
  const float* Wks = (const float*)d_in[5];  const float* bks = (const float*)d_in[6];
  const float* Wvs = (const float*)d_in[7];  const float* bvs = (const float*)d_in[8];
  const float* Wqe = (const float*)d_in[9];  const float* bqe = (const float*)d_in[10];
  const float* Wke = (const float*)d_in[11]; const float* bke = (const float*)d_in[12];
  const float* Wve = (const float*)d_in[13]; const float* bve = (const float*)d_in[14];
  const float* W1 = (const float*)d_in[15];  const float* b1 = (const float*)d_in[16];
  const float* W2 = (const float*)d_in[17];  const float* b2 = (const float*)d_in[18];
  const float* gamma = (const float*)d_in[19];
  const float* ew    = (const float*)d_in[20];

  const size_t R = (size_t)BB * NN;  // 16384 rows on both q and kv side
  char* wsb = (char*)d_ws;
  size_t off = 0;
  auto carve = [&](size_t bytes) -> void* {
    void* p = wsb + off;
    off += (bytes + 255) & ~(size_t)255;
    return p;
  };
  bf16* qbf = (bf16*)carve(R * DIN * sizeof(bf16));
  bf16* vbf = (bf16*)carve(R * DIN * sizeof(bf16));
  bf16* WTqs = (bf16*)carve((size_t)DIN * HH * sizeof(bf16));
  bf16* WTqe = (bf16*)carve((size_t)DIN * HH * sizeof(bf16));
  bf16* WTks = (bf16*)carve((size_t)DIN * HH * sizeof(bf16));
  bf16* WTke = (bf16*)carve((size_t)DIN * HH * sizeof(bf16));
  bf16* WTvs = (bf16*)carve((size_t)DIN * HH * sizeof(bf16));
  bf16* WTve = (bf16*)carve((size_t)DIN * HH * sizeof(bf16));
  bf16* W1T  = (bf16*)carve((size_t)4 * HH * HH * sizeof(bf16));
  bf16* W2T  = (bf16*)carve((size_t)HH * HH * sizeof(bf16));
  bf16* qs = (bf16*)carve(R * HH * sizeof(bf16));
  bf16* qe = (bf16*)carve(R * HH * sizeof(bf16));
  bf16* ks = (bf16*)carve(R * HH * sizeof(bf16));
  bf16* ke = (bf16*)carve(R * HH * sizeof(bf16));
  bf16* vTs = (bf16*)carve(R * HH * sizeof(bf16));
  bf16* vTe = (bf16*)carve(R * HH * sizeof(bf16));
  float* q2   = (float*)carve(R * sizeof(float));
  float* k2   = (float*)carve(R * sizeof(float));
  float* dsum = (float*)carve(BB * sizeof(float));
  float* cs   = (float*)carve(R * HH * sizeof(float));
  float* ce   = (float*)carve(R * HH * sizeof(float));

  // --- prep: convert inputs + transpose/convert weights ---
  {
    size_t n = R * DIN;
    dim3 cg((unsigned)(n / (4 * 256))), cb(256);
    hipLaunchKernelGGL(cvt_bf16_kernel, cg, cb, 0, stream, q_fp,  qbf, n);
    hipLaunchKernelGGL(cvt_bf16_kernel, cg, cb, 0, stream, v_ret, vbf, n);
  }
  {
    int kn = DIN * HH;
    dim3 tg((unsigned)((kn + 255) / 256)), tb(256);
    hipLaunchKernelGGL(transpose_w_kernel, tg, tb, 0, stream, Wqs, WTqs, DIN, HH);
    hipLaunchKernelGGL(transpose_w_kernel, tg, tb, 0, stream, Wqe, WTqe, DIN, HH);
    hipLaunchKernelGGL(transpose_w_kernel, tg, tb, 0, stream, Wks, WTks, DIN, HH);
    hipLaunchKernelGGL(transpose_w_kernel, tg, tb, 0, stream, Wke, WTke, DIN, HH);
    hipLaunchKernelGGL(transpose_w_kernel, tg, tb, 0, stream, Wvs, WTvs, DIN, HH);
    hipLaunchKernelGGL(transpose_w_kernel, tg, tb, 0, stream, Wve, WTve, DIN, HH);
    int kn1 = 4 * HH * HH;
    dim3 tg1((unsigned)((kn1 + 255) / 256));
    hipLaunchKernelGGL(transpose_w_kernel, tg1, tb, 0, stream, W1, W1T, 4 * HH, HH);
    int kn2 = HH * HH;
    dim3 tg2((unsigned)((kn2 + 255) / 256));
    hipLaunchKernelGGL(transpose_w_kernel, tg2, tb, 0, stream, W2, W2T, HH, HH);
  }

  // --- projections ---
  dim3 pb(256), pg((unsigned)(R / 16));
  hipLaunchKernelGGL(proj2_kernel, pg, pb, 0, stream,
                     qbf, WTqs, bqs, WTqe, bqe, qs, qe, q2);
  hipLaunchKernelGGL(proj4_kernel, pg, pb, 0, stream,
                     vbf, WTks, bks, WTke, bke, WTvs, bvs, WTve, bve,
                     ks, ke, vTs, vTe, k2);

  // --- electronic scale (global mean of distances) ---
  hipMemsetAsync(dsum, 0, BB * sizeof(float), stream);
  dim3 ag(NN / (16 * AW), BB), ab(AW * 32);
  hipLaunchKernelGGL(dist_sum_kernel, ag, ab, 0, stream, qe, ke, q2, k2, dsum);

  // --- fused flash attention ---
  hipLaunchKernelGGL(attn_kernel, ag, ab, 0, stream,
                     qs, qe, ks, ke, vTs, vTe, q2, k2, pi, dsum, gamma, ew, cs, ce);

  // --- MLP head ---
  dim3 mg((unsigned)(R / (16 * MW))), mb(MW * 32);
  hipLaunchKernelGGL(mlp_kernel, mg, mb, 0, stream, cs, ce, W1T, b1, W2T, b2, (float*)d_out);
}
